// TransformerBlock_5411658793117
// MI455X (gfx1250) — matrix operand order repaired and checked by reading, not compiled, not run on hardware
//
#include <hip/hip_runtime.h>
#include <hip/hip_bf16.h>
#include <math.h>

typedef __attribute__((ext_vector_type(16))) __bf16          v16bf;
typedef __attribute__((ext_vector_type(8)))  float           v8f;
typedef __attribute__((ext_vector_type(8)))  unsigned short  ushort8;
typedef __attribute__((ext_vector_type(16))) unsigned short  ushort16;
typedef __attribute__((ext_vector_type(4)))  unsigned int    uint4v;
typedef __attribute__((__vector_size__(4 * sizeof(int)))) int v4i;

#define DMODEL 2048
#define NHEADS 16
#define DKH    128
#define DFF    8192
#define BATCH  2
#define SEQ    2048
#define MROWS  (BATCH * SEQ)

// ------------------------------------------------ CDNA5 async LDS DMA path
#if defined(__gfx1250__) && \
    __has_builtin(__builtin_amdgcn_global_load_async_to_lds_b128) && \
    __has_builtin(__builtin_amdgcn_s_wait_asynccnt)
#define HAVE_ASYNC_LDS 1
#else
#define HAVE_ASYNC_LDS 0
#endif

#if HAVE_ASYNC_LDS
typedef __attribute__((address_space(1))) v4i* as1_v4i;
typedef __attribute__((address_space(3))) v4i* as3_v4i;
__device__ inline void cp_async16(void* lds, const void* g) {
  __builtin_amdgcn_global_load_async_to_lds_b128((as1_v4i)g, (as3_v4i)lds, 0, 0);
}
__device__ inline void cp_async_commit_wait() {
  __builtin_amdgcn_s_wait_asynccnt(0);
}
#else
__device__ inline void cp_async16(void* lds, const void* g) {
  *(uint4v*)lds = *(const uint4v*)g;
}
__device__ inline void cp_async_commit_wait() {}
#endif

// ---------------------------------------------------------------- helpers

__device__ inline unsigned short f2bfbits(float f) {
  unsigned u = __builtin_bit_cast(unsigned, f);
  unsigned r = u + 0x7FFFu + ((u >> 16) & 1u);
  return (unsigned short)(r >> 16);
}

__device__ inline v8f zero8() {
  v8f z;
#pragma unroll
  for (int i = 0; i < 8; ++i) z[i] = 0.0f;
  return z;
}

// A fragment: 16x32 bf16 (ISA 7.12.2).  lane = (khalf<<4)|m.
__device__ inline v16bf make_afrag(const unsigned short* p0, const unsigned short* p1) {
  ushort8 lo = *(const ushort8*)p0;
  ushort8 hi = *(const ushort8*)p1;
  ushort16 t;
#pragma unroll
  for (int i = 0; i < 8; ++i) { t[i] = lo[i]; t[i + 8] = hi[i]; }
  return __builtin_bit_cast(v16bf, t);
}

// B fragment: 32x16 bf16.  lane = (khalf<<4)|n, elements: K = 8*khalf+i (i<8), 16+8*khalf+(i-8) (i>=8).
__device__ inline v16bf make_bfrag(const unsigned short* p, int ld) {
  ushort16 t;
#pragma unroll
  for (int i = 0; i < 8; ++i) { t[i] = p[i * ld]; t[i + 8] = p[(16 + i) * ld]; }
  return __builtin_bit_cast(v16bf, t);
}

__device__ inline v8f wmma_bf16(v16bf a, v16bf b, v8f c) {
  return __builtin_amdgcn_wmma_f32_16x16x32_bf16(false, a, false, b, (short)0, c,
                                                 false, false);
}

// ---------------------------------------------------------------- convert

__global__ void conv_f32_bf16(const float* __restrict__ src,
                              unsigned short* __restrict__ dst, long long n) {
  long long i = (long long)blockIdx.x * blockDim.x + threadIdx.x;
  long long stride = (long long)gridDim.x * blockDim.x;
  for (; i < n; i += stride) dst[i] = f2bfbits(src[i]);
}

// ---------------------------------------------------------------- GEMM
// C[M,N] = A[M,K](bf16) @ B[K,N](bf16) + bias[N] (+resid).  Row-major.
// Block tile 128x256xK32, 8 waves (2 M x 4 N), wave tile 64x64 (16 WMMA frags).
// MODE 0: out bf16 = acc+bias ; MODE 1: out bf16 = gelu(acc+bias)
// MODE 2: out f32  = acc+bias+resid
template <int MODE>
__global__ __launch_bounds__(256) void gemm_bf16_kernel(
    const unsigned short* __restrict__ A, const unsigned short* __restrict__ Bw,
    const float* __restrict__ bias, const float* __restrict__ resid,
    void* __restrict__ outv, int Mn, int Nn, int Kn) {
  __shared__ __align__(16) unsigned short sA[128 * 32];   //  8 KB
  __shared__ __align__(16) unsigned short sB[32 * 256];   // 16 KB

  const int t = threadIdx.x, lane = t & 31, w = t >> 5;
  const int wm = w >> 2, wn = w & 3;           // 2 x 4 waves; wave tile 64x64
  const int mblk = blockIdx.y, nblk = blockIdx.x;

  v8f acc[4][4];
#pragma unroll
  for (int mi = 0; mi < 4; ++mi)
#pragma unroll
    for (int ni = 0; ni < 4; ++ni) acc[mi][ni] = zero8();

  const int arow = t >> 1, acol = (t & 1) * 16;   // A tile 128x32 (2 x 16B/thread)
  const int brow = t >> 3, bcol = (t & 7) * 32;   // B tile 32x256 (4 x 16B/thread)
  const size_t Abase = (size_t)(mblk * 128 + arow) * Kn + acol;
  const size_t Bcol0 = (size_t)nblk * 256 + bcol;

  for (int kt = 0; kt < Kn; kt += 32) {
    __syncthreads();
    const unsigned short* ag = A + Abase + kt;
    cp_async16(&sA[arow * 32 + acol],     ag);
    cp_async16(&sA[arow * 32 + acol + 8], ag + 8);
    const unsigned short* bg = Bw + (size_t)(kt + brow) * Nn + Bcol0;
#pragma unroll
    for (int i = 0; i < 4; ++i)
      cp_async16(&sB[brow * 256 + bcol + i * 8], bg + i * 8);
    cp_async_commit_wait();
    __syncthreads();

    const int mr = lane & 15, kb = (lane >> 4) * 8;
    v16bf aF[4], bF[4];
#pragma unroll
    for (int mi = 0; mi < 4; ++mi) {
      const int m = wm * 64 + mi * 16 + mr;
      aF[mi] = make_afrag(&sA[m * 32 + kb], &sA[m * 32 + 16 + kb]);
    }
#pragma unroll
    for (int ni = 0; ni < 4; ++ni)
      bF[ni] = make_bfrag(&sB[((lane >> 4) * 8) * 256 + wn * 64 + ni * 16 + (lane & 15)], 256);
#pragma unroll
    for (int mi = 0; mi < 4; ++mi)
#pragma unroll
      for (int ni = 0; ni < 4; ++ni) acc[mi][ni] = wmma_bf16(aF[mi], bF[ni], acc[mi][ni]);
  }

  // epilogue: lane holds rows (lane>>4)*8 + r, col (lane&15)
#pragma unroll
  for (int mi = 0; mi < 4; ++mi) {
#pragma unroll
    for (int ni = 0; ni < 4; ++ni) {
      const int rowb = mblk * 128 + wm * 64 + mi * 16 + (lane >> 4) * 8;
      const int col = nblk * 256 + wn * 64 + ni * 16 + (lane & 15);
      const float bv = bias[col];
#pragma unroll
      for (int r = 0; r < 8; ++r) {
        const size_t idx = (size_t)(rowb + r) * Nn + col;
        float v = acc[mi][ni][r] + bv;
        if (MODE == 2) {
          v += resid[idx];
          ((float*)outv)[idx] = v;
        } else if (MODE == 1) {
          v = 0.5f * v * (1.0f + erff(v * 0.70710678118654752f));
          ((unsigned short*)outv)[idx] = f2bfbits(v);
        } else {
          ((unsigned short*)outv)[idx] = f2bfbits(v);
        }
      }
    }
  }
}

// ---------------------------------------------------------------- attention
// Flash attention, one block per (batch, head, 128-query tile). 8 waves,
// wave w owns query rows w*16..w*16+15 -> softmax stats stay inside the wave.
__global__ __launch_bounds__(256) void attention_kernel(
    const unsigned short* __restrict__ Qg, const unsigned short* __restrict__ Kg,
    const unsigned short* __restrict__ Vg, unsigned short* __restrict__ Og) {
  extern __shared__ __align__(16) unsigned short smem[];
  unsigned short* sQP = smem;                 // 128x128: Q tile, reused for P
  unsigned short* sK  = smem + 128 * 128;     // K^T tile  [d][t]
  unsigned short* sV  = smem + 2 * 128 * 128; // V tile    [t][d]

  const int t = threadIdx.x, lane = t & 31, w = t >> 5;
  const int nqb = SEQ / 128;
  int bid = blockIdx.x;
  const int qblk = bid % nqb; bid /= nqb;
  const int h = bid % NHEADS;
  const int bb = bid / NHEADS;
  const int q0 = qblk * 128;
  const size_t hoff = (size_t)h * DKH;
  const size_t boff = (size_t)bb * SEQ * DMODEL;

  { // stage Q tile (async DMA into LDS)
    const int row = t >> 1, cb = (t & 1) * 64;
    const unsigned short* src = Qg + boff + (size_t)(q0 + row) * DMODEL + hoff + cb;
#pragma unroll
    for (int i = 0; i < 8; ++i)
      cp_async16(&sQP[row * 128 + cb + i * 8], src + i * 8);
    cp_async_commit_wait();
  }
  __syncthreads();

  v16bf qF[4];
  {
    const int m = w * 16 + (lane & 15);
    const int kb = (lane >> 4) * 8;
#pragma unroll
    for (int kc = 0; kc < 4; ++kc)
      qF[kc] = make_afrag(&sQP[m * 128 + kc * 32 + kb], &sQP[m * 128 + kc * 32 + 16 + kb]);
  }

  v8f O[8];
  float l[8], rmax[8];
#pragma unroll
  for (int i = 0; i < 8; ++i) { O[i] = zero8(); l[i] = 0.0f; rmax[i] = -1e30f; }

  const float scale = 0.0883883476483184f; // 1/sqrt(128)

  for (int j = 0; j <= qblk; ++j) {
    const int t0 = j * 128;
    __syncthreads();
    { // stage K^T (manual element transpose) and V (async DMA)
      const int tl = t >> 1, db = (t & 1) * 64;
      const unsigned short* ks = Kg + boff + (size_t)(t0 + tl) * DMODEL + hoff + db;
#pragma unroll 8
      for (int d = 0; d < 64; ++d) sK[(db + d) * 128 + tl] = ks[d];
      const unsigned short* vs = Vg + boff + (size_t)(t0 + tl) * DMODEL + hoff + db;
#pragma unroll
      for (int i = 0; i < 8; ++i)
        cp_async16(&sV[tl * 128 + db + i * 8], vs + i * 8);
      cp_async_commit_wait();
    }
    __syncthreads();

    // S = Q K^T * scale : hoist 8 B-frags per K-step, then 8 back-to-back WMMAs
    v8f sc[8];
#pragma unroll
    for (int ni = 0; ni < 8; ++ni) sc[ni] = zero8();
#pragma unroll
    for (int kc = 0; kc < 4; ++kc) {
      v16bf bK[8];
#pragma unroll
      for (int ni = 0; ni < 8; ++ni)
        bK[ni] = make_bfrag(&sK[(kc * 32 + (lane >> 4) * 8) * 128 + ni * 16 + (lane & 15)], 128);
#pragma unroll
      for (int ni = 0; ni < 8; ++ni) sc[ni] = wmma_bf16(qF[kc], bK[ni], sc[ni]);
    }
#pragma unroll
    for (int ni = 0; ni < 8; ++ni)
#pragma unroll
      for (int r = 0; r < 8; ++r) sc[ni][r] *= scale;

    if (j == qblk) { // causal mask on diagonal tile
      const int qloc = w * 16 + (lane >> 4) * 8;
      const int tcol = lane & 15;
#pragma unroll
      for (int ni = 0; ni < 8; ++ni) {
        const int tl2 = ni * 16 + tcol;
#pragma unroll
        for (int r = 0; r < 8; ++r)
          sc[ni][r] = (tl2 > qloc + r) ? -1e30f : sc[ni][r];
      }
    }

    // online softmax (per-row stats across 16 lanes of the half-wave)
    float nm[8];
#pragma unroll
    for (int r = 0; r < 8; ++r) {
      float m = sc[0][r];
#pragma unroll
      for (int ni = 1; ni < 8; ++ni) m = fmaxf(m, sc[ni][r]);
#pragma unroll
      for (int o = 1; o < 16; o <<= 1) m = fmaxf(m, __shfl_xor(m, o, 32));
      nm[r] = fmaxf(rmax[r], m);
    }
    float psum[8];
#pragma unroll
    for (int r = 0; r < 8; ++r) psum[r] = 0.0f;
#pragma unroll
    for (int ni = 0; ni < 8; ++ni)
#pragma unroll
      for (int r = 0; r < 8; ++r) {
        float p = __expf(sc[ni][r] - nm[r]);
        sc[ni][r] = p;
        psum[r] += p;
      }
#pragma unroll
    for (int r = 0; r < 8; ++r) {
#pragma unroll
      for (int o = 1; o < 16; o <<= 1) psum[r] += __shfl_xor(psum[r], o, 32);
      const float alpha = __expf(rmax[r] - nm[r]);
      l[r] = l[r] * alpha + psum[r];
      rmax[r] = nm[r];
#pragma unroll
      for (int ni = 0; ni < 8; ++ni) O[ni][r] *= alpha;
    }

    { // spill P (bf16) to LDS so it can be re-read in A-fragment layout
      const int rowb = w * 16 + (lane >> 4) * 8;
      const int col = lane & 15;
#pragma unroll
      for (int ni = 0; ni < 8; ++ni)
#pragma unroll
        for (int r = 0; r < 8; ++r)
          sQP[(rowb + r) * 128 + ni * 16 + col] = f2bfbits(sc[ni][r]);
    }
    __syncthreads();

    { // O += P @ V : hoisted B-frags, back-to-back WMMAs
      const int m = w * 16 + (lane & 15);
      const int kb = (lane >> 4) * 8;
#pragma unroll
      for (int kc = 0; kc < 4; ++kc) {
        v16bf aP = make_afrag(&sQP[m * 128 + kc * 32 + kb],
                              &sQP[m * 128 + kc * 32 + 16 + kb]);
        v16bf bV[8];
#pragma unroll
        for (int ni = 0; ni < 8; ++ni)
          bV[ni] = make_bfrag(&sV[(kc * 32 + (lane >> 4) * 8) * 128 + ni * 16 + (lane & 15)], 128);
#pragma unroll
        for (int ni = 0; ni < 8; ++ni) O[ni] = wmma_bf16(aP, bV[ni], O[ni]);
      }
    }
  }

  { // normalize + write context (bf16, head-merged layout)
    const int rowb = q0 + w * 16 + (lane >> 4) * 8;
    const int col = lane & 15;
#pragma unroll
    for (int ni = 0; ni < 8; ++ni)
#pragma unroll
      for (int r = 0; r < 8; ++r) {
        const float o = O[ni][r] / l[r];
        Og[boff + (size_t)(rowb + r) * DMODEL + hoff + ni * 16 + col] = f2bfbits(o);
      }
  }
}

// ---------------------------------------------------------------- layernorm
template <bool WF32, bool WBF16>
__global__ __launch_bounds__(256) void layernorm_kernel(
    const float* __restrict__ in, const float* __restrict__ g,
    const float* __restrict__ bta, float* __restrict__ outF,
    unsigned short* __restrict__ outB) {
  __shared__ float red[256];
  const int row = blockIdx.x, t = threadIdx.x;
  const float* x = in + (size_t)row * DMODEL;
  float loc[8];
  float s = 0.0f;
#pragma unroll
  for (int i = 0; i < 8; ++i) { loc[i] = x[t + i * 256]; s += loc[i]; }
  red[t] = s;
  __syncthreads();
  for (int o = 128; o > 0; o >>= 1) {
    if (t < o) red[t] += red[t + o];
    __syncthreads();
  }
  const float mu = red[0] * (1.0f / DMODEL);
  __syncthreads();
  float v = 0.0f;
#pragma unroll
  for (int i = 0; i < 8; ++i) { const float d = loc[i] - mu; v += d * d; }
  red[t] = v;
  __syncthreads();
  for (int o = 128; o > 0; o >>= 1) {
    if (t < o) red[t] += red[t + o];
    __syncthreads();
  }
  const float rstd = rsqrtf(red[0] * (1.0f / DMODEL) + 1e-5f);
#pragma unroll
  for (int i = 0; i < 8; ++i) {
    const int c = t + i * 256;
    const float y = (loc[i] - mu) * rstd * g[c] + bta[c];
    if (WF32)  outF[(size_t)row * DMODEL + c] = y;
    if (WBF16) outB[(size_t)row * DMODEL + c] = f2bfbits(y);
  }
}

// ---------------------------------------------------------------- launcher

extern "C" void kernel_launch(void* const* d_in, const int* in_sizes, int n_in,
                              void* d_out, int out_size, void* d_ws, size_t ws_size,
                              hipStream_t stream) {
  (void)in_sizes; (void)n_in; (void)out_size; (void)ws_size;
  const float* x    = (const float*)d_in[0];
  const float* wq   = (const float*)d_in[2];
  const float* bq   = (const float*)d_in[3];
  const float* wk   = (const float*)d_in[4];
  const float* bk   = (const float*)d_in[5];
  const float* wv   = (const float*)d_in[6];
  const float* bv   = (const float*)d_in[7];
  const float* wo   = (const float*)d_in[8];
  const float* bo   = (const float*)d_in[9];
  const float* w1   = (const float*)d_in[10];
  const float* b1   = (const float*)d_in[11];
  const float* w2   = (const float*)d_in[12];
  const float* b2   = (const float*)d_in[13];
  const float* ln1g = (const float*)d_in[14];
  const float* ln1b = (const float*)d_in[15];
  const float* ln2g = (const float*)d_in[16];
  const float* ln2b = (const float*)d_in[17];

  char* ws = (char*)d_ws;
  size_t off = 0;
  auto take = [&](size_t bytes) {
    char* p = ws + off;
    off += (bytes + 255) & ~(size_t)255;
    return p;
  };
  const size_t DD = (size_t)DMODEL * DMODEL;
  const size_t DF = (size_t)DMODEL * DFF;
  const size_t MD = (size_t)MROWS * DMODEL;
  const size_t MF = (size_t)MROWS * DFF;

  unsigned short* wqb  = (unsigned short*)take(DD * 2);
  unsigned short* wkb  = (unsigned short*)take(DD * 2);
  unsigned short* wvb  = (unsigned short*)take(DD * 2);
  unsigned short* wob  = (unsigned short*)take(DD * 2);
  unsigned short* w1b  = (unsigned short*)take(DF * 2);
  unsigned short* w2b  = (unsigned short*)take(DF * 2);
  unsigned short* xb   = (unsigned short*)take(MD * 2);
  unsigned short* qb   = (unsigned short*)take(MD * 2);
  unsigned short* kbuf = (unsigned short*)take(MD * 2);
  unsigned short* vbuf = (unsigned short*)take(MD * 2);
  unsigned short* ctxb = (unsigned short*)take(MD * 2);
  unsigned short* x1b  = (unsigned short*)take(MD * 2);
  unsigned short* hb   = (unsigned short*)take(MF * 2);
  float* y1  = (float*)take(MD * 4);   // pre-LN1 sum, reused for pre-LN2 sum
  float* x1f = (float*)take(MD * 4);   // LN1 output (f32 residual source)

  // 1. fp32 -> bf16 conversions
  conv_f32_bf16<<<1024, 256, 0, stream>>>(wq, wqb, (long long)DD);
  conv_f32_bf16<<<1024, 256, 0, stream>>>(wk, wkb, (long long)DD);
  conv_f32_bf16<<<1024, 256, 0, stream>>>(wv, wvb, (long long)DD);
  conv_f32_bf16<<<1024, 256, 0, stream>>>(wo, wob, (long long)DD);
  conv_f32_bf16<<<2048, 256, 0, stream>>>(w1, w1b, (long long)DF);
  conv_f32_bf16<<<2048, 256, 0, stream>>>(w2, w2b, (long long)DF);
  conv_f32_bf16<<<1024, 256, 0, stream>>>(x, xb, (long long)MD);

  // 2. QKV projections (bf16 WMMA GEMM, bias epilogue, bf16 out)
  dim3 gD(DMODEL / 256, MROWS / 128);
  gemm_bf16_kernel<0><<<gD, 256, 0, stream>>>(xb, wqb, bq, nullptr, qb,   MROWS, DMODEL, DMODEL);
  gemm_bf16_kernel<0><<<gD, 256, 0, stream>>>(xb, wkb, bk, nullptr, kbuf, MROWS, DMODEL, DMODEL);
  gemm_bf16_kernel<0><<<gD, 256, 0, stream>>>(xb, wvb, bv, nullptr, vbuf, MROWS, DMODEL, DMODEL);

  // 3. causal flash attention (WMMA for QK^T and PV)
  attention_kernel<<<dim3(BATCH * NHEADS * (SEQ / 128)), 256, 3 * 128 * 128 * 2, stream>>>(
      qb, kbuf, vbuf, ctxb);

  // 4. output projection + residual -> y1 (f32); LN1 -> x1f (f32) + x1b (bf16)
  gemm_bf16_kernel<2><<<gD, 256, 0, stream>>>(ctxb, wob, bo, x, y1, MROWS, DMODEL, DMODEL);
  layernorm_kernel<true, true><<<MROWS, 256, 0, stream>>>(y1, ln1g, ln1b, x1f, x1b);

  // 5. FFN: gelu(x1 @ w1 + b1) @ w2 + b2 + x1 -> y1 ; LN2 -> d_out
  dim3 gF(DFF / 256, MROWS / 128);
  gemm_bf16_kernel<1><<<gF, 256, 0, stream>>>(x1b, w1b, b1, nullptr, hb, MROWS, DFF, DMODEL);
  gemm_bf16_kernel<2><<<gD, 256, 0, stream>>>(hb, w2b, b2, x1f, y1, MROWS, DMODEL, DFF);
  layernorm_kernel<true, false><<<MROWS, 256, 0, stream>>>(y1, ln2g, ln2b, (float*)d_out, nullptr);
}
